// RansGinoGridToMesh_49744311222706
// MI455X (gfx1250) — matrix operand order repaired and checked by reading, not compiled, not run on hardware
//
#include <hip/hip_runtime.h>

// ---------------------------------------------------------------------------
// RansGinoGridToMesh — CDNA5 (gfx1250) WMMA implementation, round 3.
//
// Pipeline (bf16 WMMA, fp32 accumulation):
//   prep:   transpose all weights to bf16 [N][K]; convert x to bf16
//   sincos: Q[32768,256] bf16 positional embedding
//   G   = x @ proj_w + proj_b                         (WMMA, bf16 out)
//   GP  = G @ W1[:256,:]                              (WMMA, fp32 out)
//   QP  = Q @ W1[256:,:] + b1                         (WMMA, fp32 out)
//   fused edge kernel: per 64 edges (=8 queries):
//       h1 = gelu(GP[g]+QP[q]); h2 = gelu(h1@W2+b2); mean over 8 rows
//       -> H2m[32768,256] bf16   (h1/h2 never leave LDS/VGPRs)
//   M   = H2m @ W3 + b3                               (WMMA, bf16 out)
//   P   = gelu(M @ pred_w1 + pred_b1)                 (WMMA, fp32 out)
//   out = P @ pred_w2 + pred_b2                       (scalar, N=4)
//
// Round 3: fixed async builtin signature — it takes v4i pointers
// (addrspace(1) source, addrspace(3) LDS destination).
// ---------------------------------------------------------------------------

typedef __attribute__((ext_vector_type(16))) __bf16 v16bf;
typedef __attribute__((ext_vector_type(8)))  __bf16 v8bf;
typedef __attribute__((ext_vector_type(8)))  float  v8f;
typedef __attribute__((ext_vector_type(4)))  float  v4f;
typedef __attribute__((ext_vector_type(4)))  int    v4i;

#define NGRID 32768
#define NQ    32768
#define NEDGE (NQ * 8)
#define H     256
#define H2    512

// ---- async global->LDS staging (CDNA5) ----
#if __has_builtin(__builtin_amdgcn_global_load_async_to_lds_b128)
#define HAS_ASYNC_LDS 1
typedef __attribute__((address_space(1))) v4i as1_v4i;  // global
typedef __attribute__((address_space(3))) v4i as3_v4i;  // LDS
__device__ __forceinline__ void async_cp16(const void* g, void* l) {
  // LDS[l] = MEM[g], 16 bytes per lane, tracked by ASYNCcnt.
  __builtin_amdgcn_global_load_async_to_lds_b128(
      (as1_v4i*)g, (as3_v4i*)l, 0, 0);
}
#else
#define HAS_ASYNC_LDS 0
#endif

__device__ __forceinline__ void async_wait0() {
#if __has_builtin(__builtin_amdgcn_s_wait_asynccnt)
  __builtin_amdgcn_s_wait_asynccnt(0);
#else
  asm volatile("s_wait_asynccnt 0x0" ::: "memory");
#endif
}

__device__ __forceinline__ float gelu_f(float x) {
  return 0.5f * x * (1.0f + erff(x * 0.70710678118654752f));
}

__device__ __forceinline__ v8f wmma_bf16(v16bf a, v16bf b, v8f c) {
  // v_wmma_f32_16x16x32_bf16  D = A*B + C
  return __builtin_amdgcn_wmma_f32_16x16x32_bf16(
      false, a, false, b, (short)0, c, false, false);
}

// A-fragment (16x32 bf16, ISA 7.12.2): lanes0-15: M=lane, K = 0..7 & 16..23;
// lanes16-31: M=lane-16, K = 8..15 & 24..31.
__device__ __forceinline__ v16bf load_frag_a(const __bf16* base, int stride, int lane) {
  int m = lane & 15;
  int koff = (lane >> 4) * 8;
  const __bf16* p = base + m * stride + koff;
  v8bf lo = *(const v8bf*)(p);
  v8bf hi = *(const v8bf*)(p + 16);
  v16bf r;
#pragma unroll
  for (int i = 0; i < 8; ++i) { r[i] = lo[i]; r[i + 8] = hi[i]; }
  return r;
}

// B-fragment (32x16 bf16) loaded from transposed weights WT[N][K]:
// lanes0-15: N=lane, K = 0..7 & 16..23; lanes16-31: N=lane-16, K = 8..15 & 24..31.
__device__ __forceinline__ v16bf load_frag_b(const __bf16* base, int stride, int lane) {
  int n = lane & 15;
  int koff = (lane >> 4) * 8;
  const __bf16* p = base + n * stride + koff;
  v8bf lo = *(const v8bf*)(p);
  v8bf hi = *(const v8bf*)(p + 16);
  v16bf r;
#pragma unroll
  for (int i = 0; i < 8; ++i) { r[i] = lo[i]; r[i + 8] = hi[i]; }
  return r;
}

// ---------------- prep kernels ----------------

__global__ void transpose_to_bf16(const float* __restrict__ W,
                                  __bf16* __restrict__ WT, int K, int N) {
  int idx = blockIdx.x * blockDim.x + threadIdx.x;
  if (idx >= K * N) return;
  int n = idx / K;
  int k = idx - n * K;
  WT[idx] = (__bf16)W[(size_t)k * N + n];
}

__global__ void f32_to_bf16(const float* __restrict__ X,
                            __bf16* __restrict__ Y, int n) {
  int i = blockIdx.x * blockDim.x + threadIdx.x;
  if (i < n) Y[i] = (__bf16)X[i];
}

// sincos embedding: eff=84 per dim (42 omegas), 3 dims, 4 zero pad.
__global__ void sincos_embed_kernel(const float* __restrict__ qpos,
                                    __bf16* __restrict__ Q) {
  int qrow = blockIdx.x;
  int j = threadIdx.x;
  float val = 0.0f;
  if (j < 252) {
    int d = j / 84;
    int r = j - d * 84;
    int i = (r < 42) ? r : r - 42;
    // omega_i = 10000^(-2i/84)
    float omega = __expf(-(2.0f * (float)i / 84.0f) * 9.210340371976184f);
    float c = qpos[qrow * 3 + d] * omega;
    val = (r < 42) ? sinf(c) : cosf(c);
  }
  Q[(size_t)qrow * H + j] = (__bf16)val;
}

// ---------------- generic WMMA GEMM ----------------
// C[m0+64, n0+64] = act(A[M,K] @ Bt[N,K]^T + bias); 4 waves, each 16Mx64N.
__global__ __launch_bounds__(128)
void gemm_wmma(const __bf16* __restrict__ A, int lda,
               const __bf16* __restrict__ Bt, int ldb,
               const float* __restrict__ bias,
               float* __restrict__ Cf, __bf16* __restrict__ Cb, int ldc,
               int K, int do_gelu) {
  __shared__ __bf16 As[64 * 40];
  __shared__ __bf16 Bs[64 * 40];
  const int tid = threadIdx.x;
  const int lane = tid & 31;
  const int wave = tid >> 5;  // 0..3
  const int m0 = blockIdx.x * 64;
  const int n0 = blockIdx.y * 64;

  const int row = tid >> 1;
  const int half = (tid & 1) * 16;

  v8f acc[4] = {};
  for (int kc = 0; kc < K; kc += 32) {
#if HAS_ASYNC_LDS
    const __bf16* ag = A + (size_t)(m0 + row) * lda + kc + half;
    const __bf16* bg = Bt + (size_t)(n0 + row) * ldb + kc + half;
    __bf16* al = &As[row * 40 + half];
    __bf16* bl = &Bs[row * 40 + half];
    async_cp16(ag,     al);       // 8 bf16
    async_cp16(ag + 8, al + 8);   // next 8 bf16
    async_cp16(bg,     bl);
    async_cp16(bg + 8, bl + 8);
    async_wait0();
#else
    {  // stage A tile 64x32
      const v8bf* src = (const v8bf*)(A + (size_t)(m0 + row) * lda + kc + half);
      v8bf x0 = src[0], x1 = src[1];
      v8bf* dst = (v8bf*)(&As[row * 40 + half]);
      dst[0] = x0; dst[1] = x1;
    }
    {  // stage B tile 64x32 (from transposed weights)
      const v8bf* src = (const v8bf*)(Bt + (size_t)(n0 + row) * ldb + kc + half);
      v8bf x0 = src[0], x1 = src[1];
      v8bf* dst = (v8bf*)(&Bs[row * 40 + half]);
      dst[0] = x0; dst[1] = x1;
    }
#endif
    __syncthreads();
    v16bf af = load_frag_a(&As[(wave * 16) * 40], 40, lane);
#pragma unroll
    for (int j = 0; j < 4; ++j) {
      v16bf bf = load_frag_b(&Bs[(j * 16) * 40], 40, lane);
      acc[j] = wmma_bf16(af, bf, acc[j]);
    }
    __syncthreads();
  }

  // C layout: VGPR r, lane l -> M = r + 8*(l>>4), N = l&15
  const int mbase = m0 + wave * 16 + (lane >> 4) * 8;
  const int ncol = lane & 15;
#pragma unroll
  for (int j = 0; j < 4; ++j) {
    int n = n0 + j * 16 + ncol;
    float bval = bias ? bias[n] : 0.0f;
#pragma unroll
    for (int r = 0; r < 8; ++r) {
      float v = acc[j][r] + bval;
      if (do_gelu) v = gelu_f(v);
      size_t off = (size_t)(mbase + r) * ldc + n;
      if (Cf) Cf[off] = v;
      if (Cb) Cb[off] = (__bf16)v;
    }
  }
}

// ---------------- fused edge kernel ----------------
// 64 edges (= 8 whole queries, since query_idx = e>>3) per workgroup.
// h1 = gelu(GP[g]+QP[q]) built in LDS per 32-K chunk; h2 = gelu(h1@W2+b2)
// accumulated via WMMA; per-query mean over 8 rows reduced in registers.
__global__ __launch_bounds__(512)
void edge_msg_fused(const float* __restrict__ GP, const float* __restrict__ QP,
                    const __bf16* __restrict__ W2T, const float* __restrict__ b2,
                    const int* __restrict__ edges, __bf16* __restrict__ H2m) {
  __shared__ __bf16 As[64 * 40];    // 64 edges x 32 K
  __shared__ __bf16 Bs[256 * 40];   // 256 N x 32 K
  const int tid = threadIdx.x;
  const int lane = tid & 31;
  const int wave = tid >> 5;        // 0..15
  const int wy = wave >> 2;         // M subtile (16 edges)
  const int wx = wave & 3;          // N subtile (64 cols)
  const int e0 = blockIdx.x * 64;

  // A-build: 8 threads per edge row, 4 K-elements each
  const int arow = tid >> 3;
  const int asub = (tid & 7) * 4;
  const int e = e0 + arow;
  const int g = edges[2 * e + 1];   // grid index
  const int q = e >> 3;             // query index (repeat(arange,8))
  const float* gp = GP + (size_t)g * H2;
  const float* qp = QP + (size_t)q * H2;

  // B-copy: 2 threads per N row, 16 K-elements each
  const int brow = tid >> 1;
  const int bhalf = (tid & 1) * 16;
  const __bf16* w2r = W2T + (size_t)brow * H2;

  v8f acc[4] = {};
  for (int kc = 0; kc < H2; kc += 32) {
    // B tile: pure copy -> async path
#if HAS_ASYNC_LDS
    {
      const __bf16* bg = w2r + kc + bhalf;
      __bf16* bl = &Bs[brow * 40 + bhalf];
      async_cp16(bg,     bl);
      async_cp16(bg + 8, bl + 8);
    }
#else
    {
      const v8bf* src = (const v8bf*)(w2r + kc + bhalf);
      v8bf x0 = src[0], x1 = src[1];
      v8bf* dst = (v8bf*)(&Bs[brow * 40 + bhalf]);
      dst[0] = x0; dst[1] = x1;
    }
#endif
    // A tile: gather + add + gelu (computed, stays on VALU path)
    v4f gv = *(const v4f*)(gp + kc + asub);
    v4f qv = *(const v4f*)(qp + kc + asub);
#pragma unroll
    for (int i = 0; i < 4; ++i) {
      As[arow * 40 + asub + i] = (__bf16)gelu_f(gv[i] + qv[i]);
    }
#if HAS_ASYNC_LDS
    async_wait0();
#endif
    __syncthreads();
    v16bf af = load_frag_a(&As[(wy * 16) * 40], 40, lane);
#pragma unroll
    for (int j = 0; j < 4; ++j) {
      v16bf bf = load_frag_b(&Bs[(wx * 64 + j * 16) * 40], 40, lane);
      acc[j] = wmma_bf16(af, bf, acc[j]);
    }
    __syncthreads();
  }

  // Epilogue: bias + gelu, then mean over the 8 rows of each query.
  // Lanes 0-15 hold query (e0+wy*16)/8, lanes 16-31 the next query.
  const int qout = (e0 + wy * 16 + (lane >> 4) * 8) >> 3;
  const int ncol = lane & 15;
#pragma unroll
  for (int j = 0; j < 4; ++j) {
    int n = wx * 64 + j * 16 + ncol;
    float bv = b2[n];
    float s = 0.0f;
#pragma unroll
    for (int r = 0; r < 8; ++r) s += gelu_f(acc[j][r] + bv);
    H2m[(size_t)qout * H + n] = (__bf16)(s * 0.125f);
  }
}

// ---------------- tiny pred head (N=4) ----------------
__global__ void pred_head(const float* __restrict__ P,
                          const float* __restrict__ W2,
                          const float* __restrict__ b2,
                          float* __restrict__ out) {
  int row = blockIdx.x * blockDim.x + threadIdx.x;
  if (row >= NQ) return;
  v4f acc = *(const v4f*)b2;
  const float* pr = P + (size_t)row * H;
  for (int k = 0; k < H; ++k) {
    float p = pr[k];
    v4f w = *(const v4f*)(W2 + k * 4);
#pragma unroll
    for (int i = 0; i < 4; ++i) acc[i] += p * w[i];
  }
  *(v4f*)(out + (size_t)row * 4) = acc;
}

// ---------------------------------------------------------------------------

extern "C" void kernel_launch(void* const* d_in, const int* in_sizes, int n_in,
                              void* d_out, int out_size, void* d_ws, size_t ws_size,
                              hipStream_t stream) {
  const float* x         = (const float*)d_in[0];
  const float* query_pos = (const float*)d_in[1];
  const int*   edges     = (const int*)d_in[2];
  const float* proj_w    = (const float*)d_in[3];
  const float* proj_b    = (const float*)d_in[4];
  const float* msg_w1    = (const float*)d_in[5];
  const float* msg_b1    = (const float*)d_in[6];
  const float* msg_w2    = (const float*)d_in[7];
  const float* msg_b2    = (const float*)d_in[8];
  const float* msg_w3    = (const float*)d_in[9];
  const float* msg_b3    = (const float*)d_in[10];
  const float* pred_w1   = (const float*)d_in[11];
  const float* pred_b1   = (const float*)d_in[12];
  const float* pred_w2   = (const float*)d_in[13];
  const float* pred_b2   = (const float*)d_in[14];
  float* out = (float*)d_out;

  char* w = (char*)d_ws;
  auto alloc = [&](size_t bytes) {
    void* p = (void*)w;
    w += (bytes + 255) & ~(size_t)255;
    return p;
  };

  __bf16* ProjT = (__bf16*)alloc((size_t)H * H * 2);
  __bf16* W1T   = (__bf16*)alloc((size_t)H2 * H2 * 2);
  __bf16* W2T   = (__bf16*)alloc((size_t)H * H2 * 2);
  __bf16* W3T   = (__bf16*)alloc((size_t)H * H * 2);
  __bf16* P1T   = (__bf16*)alloc((size_t)H * H * 2);
  __bf16* Xbf   = (__bf16*)alloc((size_t)NGRID * H * 2);
  __bf16* Qbf   = (__bf16*)alloc((size_t)NQ * H * 2);
  __bf16* Gbf   = (__bf16*)alloc((size_t)NGRID * H * 2);
  float*  GP    = (float*)alloc((size_t)NGRID * H2 * 4);
  float*  QP    = (float*)alloc((size_t)NQ * H2 * 4);
  __bf16* H2m   = (__bf16*)alloc((size_t)NQ * H * 2);
  __bf16* Mbf   = (__bf16*)alloc((size_t)NQ * H * 2);
  float*  Pf    = (float*)alloc((size_t)NQ * H * 4);

  // --- weight prep (transpose to [N][K] bf16) ---
  transpose_to_bf16<<<(H * H + 255) / 256, 256, 0, stream>>>(proj_w, ProjT, H, H);
  transpose_to_bf16<<<(H2 * H2 + 255) / 256, 256, 0, stream>>>(msg_w1, W1T, H2, H2);
  transpose_to_bf16<<<(H2 * H + 255) / 256, 256, 0, stream>>>(msg_w2, W2T, H2, H);
  transpose_to_bf16<<<(H * H + 255) / 256, 256, 0, stream>>>(msg_w3, W3T, H, H);
  transpose_to_bf16<<<(H * H + 255) / 256, 256, 0, stream>>>(pred_w1, P1T, H, H);
  f32_to_bf16<<<(NGRID * H + 255) / 256, 256, 0, stream>>>(x, Xbf, NGRID * H);
  sincos_embed_kernel<<<NQ, H, 0, stream>>>(query_pos, Qbf);

  // --- G = x @ proj_w + proj_b  (bf16 out) ---
  gemm_wmma<<<dim3(NGRID / 64, H / 64), 128, 0, stream>>>(
      Xbf, H, ProjT, H, proj_b, nullptr, Gbf, H, H, 0);

  // --- GP = G @ W1[:256,:]  (fp32 out, no bias) ---
  gemm_wmma<<<dim3(NGRID / 64, H2 / 64), 128, 0, stream>>>(
      Gbf, H, W1T, H2, nullptr, GP, nullptr, H2, H, 0);

  // --- QP = Q @ W1[256:,:] + b1  (fp32 out) ---
  gemm_wmma<<<dim3(NQ / 64, H2 / 64), 128, 0, stream>>>(
      Qbf, H, W1T + H, H2, msg_b1, QP, nullptr, H2, H, 0);

  // --- fused edge MLP layer2 + per-query mean ---
  edge_msg_fused<<<NEDGE / 64, 512, 0, stream>>>(GP, QP, W2T, msg_b2, edges, H2m);

  // --- M = H2m @ W3 + b3  (bf16 out) ---
  gemm_wmma<<<dim3(NQ / 64, H / 64), 128, 0, stream>>>(
      H2m, H, W3T, H, msg_b3, nullptr, Mbf, H, H, 0);

  // --- P = gelu(M @ pred_w1 + pred_b1)  (fp32 out) ---
  gemm_wmma<<<dim3(NQ / 64, H / 64), 128, 0, stream>>>(
      Mbf, H, P1T, H, pred_b1, Pf, nullptr, H, H, 1);

  // --- out = P @ pred_w2 + pred_b2 ---
  pred_head<<<(NQ + 255) / 256, 256, 0, stream>>>(Pf, pred_w2, pred_b2, out);
}
